// SelfAttentionHead_63745904607500
// MI455X (gfx1250) — matrix operand order repaired and checked by reading, not compiled, not run on hardware
//
#include <hip/hip_runtime.h>
#include <hip/hip_bf16.h>

// ---------------------------------------------------------------------------
// SelfAttentionHead: B=8, T=2048, C=1024, H=128, causal, fp32 in/out.
// Strategy:
//   0) bandwidth-bound fp32->bf16 pre-convert of x and Wq/Wk/Wv (each value
//      converted exactly once; keeps v_cvt out of the GEMM inner loops)
//   1) QKV projection with v_wmma_f32_16x16x32_bf16 (pure b128-load + wmma loop)
//   2) causal flash attention (QK^T + PV via WMMA, fp32 online softmax)
// ---------------------------------------------------------------------------

typedef __bf16 bf16_t;
typedef __attribute__((ext_vector_type(16))) __bf16 v16bf;
typedef __attribute__((ext_vector_type(8)))  __bf16 v8bf;
typedef __attribute__((ext_vector_type(8)))  float  v8f;

static constexpr int BB = 8;
static constexpr int TT = 2048;
static constexpr int CC = 1024;
static constexpr int HH = 128;

#define WMMA_BF16(a, b, c) \
  __builtin_amdgcn_wmma_f32_16x16x32_bf16(false, (a), false, (b), (short)0, (c), false, false)

__device__ __forceinline__ v16bf combine16(v8bf lo, v8bf hi) {
  return __builtin_shufflevector(lo, hi, 0, 1, 2, 3, 4, 5, 6, 7,
                                         8, 9, 10, 11, 12, 13, 14, 15);
}

// Load 16 contiguous bf16 (32B, two b128s) -> fragment half-pair.
__device__ __forceinline__ v16bf load16bf(const bf16_t* __restrict__ p) {
  return combine16(*(const v8bf*)p, *(const v8bf*)(p + 8));
}

// ---------------------------------------------------------------------------
// Kernel 0: fp32 -> bf16 convert, 8 elements/thread (b128 in, b128 out).
// ---------------------------------------------------------------------------
__global__ __launch_bounds__(256) void attn_cvt_bf16_kernel(
    const float* __restrict__ src, bf16_t* __restrict__ dst, int n8) {
  const int i = blockIdx.x * 256 + threadIdx.x;
  if (i >= n8) return;
  const float4 a = *(const float4*)(src + (size_t)i * 8);
  const float4 b = *(const float4*)(src + (size_t)i * 8 + 4);
  v8bf r;
  r[0] = (__bf16)a.x; r[1] = (__bf16)a.y; r[2] = (__bf16)a.z; r[3] = (__bf16)a.w;
  r[4] = (__bf16)b.x; r[5] = (__bf16)b.y; r[6] = (__bf16)b.z; r[7] = (__bf16)b.w;
  *(v8bf*)(dst + (size_t)i * 8) = r;
}

// ---------------------------------------------------------------------------
// Kernel 1: fused QKV projection (all-bf16 operands).
//   q[b,t,h] = sum_c x[b,t,c] * Wq[h,c]   (same for k)
//   vT[b,h,t] = sum_c x[b,t,c] * Wv[h,c]  (V stored transposed for PV GEMM)
// Grid: B*(T/16) blocks, 256 threads (8 waves). Wave w owns h-tile [16w,16w+16).
// Inner loop per 32-deep K-chunk: 8x global_load_b128 + 3x v_wmma. No VALU cvt.
// ---------------------------------------------------------------------------
__global__ __launch_bounds__(256) void attn_qkv_proj_kernel(
    const bf16_t* __restrict__ x,
    const bf16_t* __restrict__ Wq,
    const bf16_t* __restrict__ Wk,
    const bf16_t* __restrict__ Wv,
    bf16_t* __restrict__ qo, bf16_t* __restrict__ ko, bf16_t* __restrict__ vT) {
  const int lane = threadIdx.x & 31;
  const int wave = threadIdx.x >> 5;
  const int b    = blockIdx.x >> 7;     // / (T/16) == /128
  const int it   = blockIdx.x & 127;
  const int t0   = it << 4;
  const int hb   = wave << 4;           // h-tile base
  const int ln   = lane & 15;
  const int hi   = lane >> 4;           // lane half (0/1)

  v8f accq, acck, accv;
#pragma unroll
  for (int i = 0; i < 8; ++i) { accq[i] = 0.f; acck[i] = 0.f; accv[i] = 0.f; }

  const bf16_t* xrow = x  + (size_t)(b * TT + t0 + ln) * CC;   // A: lane = row
  const bf16_t* wqr  = Wq + (size_t)(hb + ln) * CC;            // B: lane = col (h)
  const bf16_t* wkr  = Wk + (size_t)(hb + ln) * CC;
  const bf16_t* wvr  = Wv + (size_t)(hb + ln) * CC;

  for (int cb = 0; cb < CC; cb += 32) {
    // A fragment (16x32): lanes 0-15 K=0..15, lanes 16-31 K=16..31 (contiguous)
    const bf16_t* pa = xrow + cb + hi * 16;
    v16bf af = combine16(*(const v8bf*)pa, *(const v8bf*)(pa + 8));
    // B fragments (32x16): lanes 0-15 K=0..15, lanes 16-31 K=16..31 (contiguous)
    v16bf bq = load16bf(wqr + cb + hi * 16);
    v16bf bk = load16bf(wkr + cb + hi * 16);
    v16bf bv = load16bf(wvr + cb + hi * 16);
    accq = WMMA_BF16(af, bq, accq);
    acck = WMMA_BF16(af, bk, acck);
    accv = WMMA_BF16(af, bv, accv);
  }

  // D layout: VGPR r -> row (r + 8*hi), col = ln.
#pragma unroll
  for (int r = 0; r < 8; ++r) {
    const int m = r + hi * 8;
    const int t = t0 + m;
    qo[(size_t)(b * TT + t) * HH + hb + ln] = (__bf16)accq[r];
    ko[(size_t)(b * TT + t) * HH + hb + ln] = (__bf16)acck[r];
    vT[((size_t)b * HH + hb + ln) * TT + t] = (__bf16)accv[r];
  }
}

// ---------------------------------------------------------------------------
// Kernel 2: causal flash attention over bf16 q/k/vT, fp32 accumulate+softmax.
// Grid: B*(T/16)/2 blocks of 64 threads; each wave independently owns one
// 16-query-row tile (no cross-wave sync -> causal trip-count divergence OK).
// Key blocks of 32 -> QK^T: 8 WMMAs, PV: 8 WMMAs per block.
// ---------------------------------------------------------------------------
__global__ __launch_bounds__(64) void attn_flash_kernel(
    const bf16_t* __restrict__ q, const bf16_t* __restrict__ k,
    const bf16_t* __restrict__ vT, float* __restrict__ out) {
  const int lane = threadIdx.x & 31;
  const int wave = threadIdx.x >> 5;
  const int g    = blockIdx.x * 2 + wave;   // global q-tile id
  const int b    = g >> 7;
  const int it   = g & 127;
  const int t0   = it << 4;
  const int ln   = lane & 15;
  const int hi   = lane >> 4;

  // Padded P-transpose staging (row stride 40 bf16 = 80B: 16B-aligned rows,
  // conflict-free strided reads). 1280B per wave.
  __shared__ __align__(16) __bf16 plds_all[2 * 16 * 40];
  __bf16* pl = plds_all + wave * (16 * 40);

  // Q fragments: 4 chunks of K=32 over the H=128 contraction.
  v16bf qf[4];
  const bf16_t* qrow = q + (size_t)(b * TT + t0 + ln) * HH;
#pragma unroll
  for (int cc = 0; cc < 4; ++cc) {
    const int cb = cc * 32;
    qf[cc] = combine16(*(const v8bf*)(qrow + cb + hi * 16),
                       *(const v8bf*)(qrow + cb + 8 + hi * 16));
  }

  float m_i[8], l_i[8];
  v8f acc[8];
#pragma unroll
  for (int r = 0; r < 8; ++r) { m_i[r] = -3.0e38f; l_i[r] = 0.f; }
#pragma unroll
  for (int n = 0; n < 8; ++n)
#pragma unroll
    for (int i = 0; i < 8; ++i) acc[n][i] = 0.f;

  const float scale = 0.08838834764831845f;  // 1/sqrt(128)
  const int nblocks = (t0 + 47) >> 5;        // key blocks of 32, causal bound

  for (int jb = 0; jb < nblocks; ++jb) {
    const int kb0 = jb << 5;

    if (jb + 1 < nblocks) {  // prefetch next key block rows
      __builtin_prefetch(k + (size_t)(b * TT + kb0 + 32 + ln) * HH, 0, 1);
    }

    // ---- S = Q @ K^T (16x32 scores, two 16-col N-tiles) ----
    v8f s0, s1;
#pragma unroll
    for (int i = 0; i < 8; ++i) { s0[i] = 0.f; s1[i] = 0.f; }
#pragma unroll
    for (int cc = 0; cc < 4; ++cc) {
      const int cb = cc * 32;
      const bf16_t* kr0 = k + (size_t)(b * TT + kb0 + ln) * HH + cb + hi * 16;
      const bf16_t* kr1 = k + (size_t)(b * TT + kb0 + 16 + ln) * HH + cb + hi * 16;
      s0 = WMMA_BF16(qf[cc], load16bf(kr0), s0);
      s1 = WMMA_BF16(qf[cc], load16bf(kr1), s1);
    }

    // ---- online softmax (rows live across 16-lane halves) ----
#pragma unroll
    for (int r = 0; r < 8; ++r) {
      const int m = r + hi * 8;
      const int t = t0 + m;
      const int k0i = kb0 + ln;
      float x0 = (k0i <= t)      ? s0[r] * scale : -3.0e38f;
      float x1 = (k0i + 16 <= t) ? s1[r] * scale : -3.0e38f;
      float mx = fmaxf(x0, x1);
#pragma unroll
      for (int off = 1; off < 16; off <<= 1) mx = fmaxf(mx, __shfl_xor(mx, off, 16));
      const float mn   = fmaxf(m_i[r], mx);
      const float corr = __expf(m_i[r] - mn);
      const float p0   = __expf(x0 - mn);
      const float p1   = __expf(x1 - mn);
      float rs = p0 + p1;
#pragma unroll
      for (int off = 1; off < 16; off <<= 1) rs += __shfl_xor(rs, off, 16);
      m_i[r] = mn;
      l_i[r] = l_i[r] * corr + rs;
#pragma unroll
      for (int n = 0; n < 8; ++n) acc[n][r] *= corr;
      // stage P (D layout) into LDS for the A-layout transpose
      pl[m * 40 + ln]      = (__bf16)p0;
      pl[m * 40 + 16 + ln] = (__bf16)p1;
    }

    // cross-lane LDS RAW within the wave: explicit DS-counter fence
    asm volatile("s_wait_dscnt 0" ::: "memory");

    // ---- O += P @ V : read P back in A-matrix striping, V from vT ----
    {
      const int base = ln * 40 + hi * 16;
      v16bf pf = combine16(*(const v8bf*)(pl + base),
                           *(const v8bf*)(pl + base + 8));
#pragma unroll
      for (int n = 0; n < 8; ++n) {
        const bf16_t* vr = vT + ((size_t)b * HH + n * 16 + ln) * TT + kb0 + hi * 16;
        acc[n] = WMMA_BF16(pf, load16bf(vr), acc[n]);
      }
    }
  }

  // ---- epilogue: normalize and store fp32 ----
#pragma unroll
  for (int r = 0; r < 8; ++r) {
    const int m = r + hi * 8;
    const float inv = 1.0f / l_i[r];
    float* orow = out + (size_t)(b * TT + t0 + m) * HH;
#pragma unroll
    for (int n = 0; n < 8; ++n) orow[n * 16 + ln] = acc[n][r] * inv;
  }
}

// ---------------------------------------------------------------------------
// Launch
// ---------------------------------------------------------------------------
extern "C" void kernel_launch(void* const* d_in, const int* in_sizes, int n_in,
                              void* d_out, int out_size, void* d_ws, size_t ws_size,
                              hipStream_t stream) {
  const float* x  = (const float*)d_in[0];
  const float* Wq = (const float*)d_in[1];
  const float* Wk = (const float*)d_in[2];
  const float* Wv = (const float*)d_in[3];
  float* out = (float*)d_out;

  // Workspace layout (bf16):
  //   xb  [B*T*C]  33.5 MB
  //   wqb/wkb/wvb [H*C] 256 KB each
  //   q, k [B*T*H], vT [B*H*T]  4 MB each
  const size_t nBTC = (size_t)BB * TT * CC;
  const size_t nHC  = (size_t)HH * CC;
  const size_t nBTH = (size_t)BB * TT * HH;
  bf16_t* xb  = (bf16_t*)d_ws;
  bf16_t* wqb = xb  + nBTC;
  bf16_t* wkb = wqb + nHC;
  bf16_t* wvb = wkb + nHC;
  bf16_t* qws = wvb + nHC;
  bf16_t* kws = qws + nBTH;
  bf16_t* vTw = kws + nBTH;

  // 0) pre-convert fp32 -> bf16 (each value converted exactly once)
  {
    const int n8x = (int)(nBTC / 8);
    attn_cvt_bf16_kernel<<<(n8x + 255) / 256, 256, 0, stream>>>(x, xb, n8x);
    const int n8w = (int)(nHC / 8);
    attn_cvt_bf16_kernel<<<(n8w + 255) / 256, 256, 0, stream>>>(Wq, wqb, n8w);
    attn_cvt_bf16_kernel<<<(n8w + 255) / 256, 256, 0, stream>>>(Wk, wkb, n8w);
    attn_cvt_bf16_kernel<<<(n8w + 255) / 256, 256, 0, stream>>>(Wv, wvb, n8w);
  }

  // 1) QKV projection (WMMA), V stored transposed
  attn_qkv_proj_kernel<<<BB * (TT / 16), 256, 0, stream>>>(xb, wqb, wkb, wvb,
                                                           qws, kws, vTw);
  // 2) causal flash attention (WMMA)
  attn_flash_kernel<<<BB * (TT / 16) / 2, 64, 0, stream>>>(qws, kws, vTw, out);
}
